// Transformer_51900384805045
// MI455X (gfx1250) — matrix operand order repaired and checked by reading, not compiled, not run on hardware
//
#include <hip/hip_runtime.h>

// ---------------------------------------------------------------------------
// Transformer forward on gfx1250 (MI455X).
// All GEMMs: v_wmma_f32_16x16x32_f16, double-buffered LDS tiles filled with
// GLOBAL_LOAD_ASYNC_TO_LDS_B128 (ASYNCcnt), one barrier per K-step.
// Every B operand is K-contiguous (weights pre-transposed, V written
// transposed), so all LDS traffic is b128.
// ---------------------------------------------------------------------------

#define LAYERS 6
#define DM     512
#define NH     8
#define DHD    64
#define FF     2048
#define NB     8
#define SSQ    1024
#define PADTOK 25
#define NEGINF (-1e9f)

typedef __attribute__((ext_vector_type(16))) _Float16 v16h;
typedef __attribute__((ext_vector_type(8)))  float    v8f;
typedef __attribute__((ext_vector_type(4)))  float    f32x4;

#define AS_STRIDE 40   // 128 x 32 f16 tile rows padded to 40 f16 (80B, 16B aligned)
#define BS_STRIDE 40   // 64(N) x 32(K) f16 tile, stored N-major

union FragH { v16h v; f32x4 q[2]; _Float16 e[16]; };

// ---------------------------------------------------------------------------
// gfx1250 async copy: 16B global -> LDS, tracked by ASYNCcnt.
// LDS address = low 32 bits of the generic pointer.
// ---------------------------------------------------------------------------
__device__ __forceinline__ void async_copy16(_Float16* ldst, const _Float16* gsrc)
{
    unsigned l = (unsigned)(unsigned long long)(uintptr_t)ldst;
    unsigned long long g = (unsigned long long)(uintptr_t)gsrc;
    asm volatile("global_load_async_to_lds_b128 %0, %1, off"
                 :: "v"(l), "v"(g) : "memory");
}
__device__ __forceinline__ void wait_async0()
{
    asm volatile("s_wait_asynccnt 0x0" ::: "memory");
}
__device__ __forceinline__ void wait_ds0()
{
    asm volatile("s_wait_dscnt 0x0" ::: "memory");
}

// issue one 128x32 A tile + 64x32 B tile into LDS buffers (3 async ops/thread)
__device__ __forceinline__ void issue_tile(
    const _Float16* __restrict__ A, int lda,
    const _Float16* __restrict__ Bm, int ldb,
    int m0, int n0, int k0, _Float16* As, _Float16* Bs, int t)
{
    const int ar = t >> 1;            // 0..127
    const int ac = (t & 1) * 16;      // 0 or 16
    const _Float16* ga = A + (size_t)(m0 + ar) * lda + k0 + ac;
    _Float16* la = As + ar * AS_STRIDE + ac;
    async_copy16(la,     ga);
    async_copy16(la + 8, ga + 8);
    const int bn = t >> 2;            // 0..63
    const int bk = (t & 3) * 8;       // 0,8,16,24
    async_copy16(Bs + bn * BS_STRIDE + bk,
                 Bm + (size_t)(n0 + bn) * ldb + k0 + bk);
}

// ---------------------------------------------------------------------------
// Core: block 256 threads (8 waves, 4Mx2N), block tile 128(M) x 64(N), BK=32.
// Each wave computes a 32x32 sub-tile = 2x2 WMMA accumulators.
// A row-major [M,K] stride lda. B source is N-major: B[k][n] at Bm[n*ldb+k].
// Double-buffered LDS, async fills, one barrier per K-step.
// ---------------------------------------------------------------------------
__device__ __forceinline__ void gemm_core(
    const _Float16* __restrict__ A, int lda,
    const _Float16* __restrict__ Bm, int ldb,
    int m0, int n0, int K,
    _Float16* As0, _Float16* Bs0, _Float16* As1, _Float16* Bs1,
    v8f c[2][2])
{
    const int t      = threadIdx.x;
    const int lane   = t & 31;
    const int waveId = t >> 5;
    const int wm     = (waveId >> 1) * 32;
    const int wn     = (waveId & 1) * 32;

    v8f zero = {};
    c[0][0] = zero; c[0][1] = zero; c[1][0] = zero; c[1][1] = zero;

    // fragment indices (ISA 16-bit A 16x32 / B 32x16 wave32 layouts)
    const int mrow  = lane & 15;
    const int abase = (lane < 16) ? 0 : 8;   // A: elems0..7 K=abase.., elems8..15 K=abase+16..
    const int kb    = (lane < 16) ? 0 : 8;   // B: elems0..7 K=kb.., elems8..15 K=kb+16.., col = lane&15

    issue_tile(A, lda, Bm, ldb, m0, n0, 0, As0, Bs0, t);

    _Float16* Asc = As0; _Float16* Bsc = Bs0;
    _Float16* Asn = As1; _Float16* Bsn = Bs1;

    for (int k0 = 0; k0 < K; k0 += 32) {
        wait_async0();      // current buffers filled (all my async ops done)
        wait_ds0();         // my LDS reads of the other buffer fully serviced
        __syncthreads();    // everyone done -> safe to read cur / overwrite next
        if (k0 + 32 < K)
            issue_tile(A, lda, Bm, ldb, m0, n0, k0 + 32, Asn, Bsn, t);

        FragH a0, a1, b0, b1;
        {
            const _Float16* pa = Asc + (wm + mrow) * AS_STRIDE + abase;
            a0.q[0] = *(const f32x4*)pa;
            a0.q[1] = *(const f32x4*)(pa + 16);
            pa += 16 * AS_STRIDE;
            a1.q[0] = *(const f32x4*)pa;
            a1.q[1] = *(const f32x4*)(pa + 16);
        }
        {
            const _Float16* pb = Bsc + (wn + mrow) * BS_STRIDE + kb;
            b0.q[0] = *(const f32x4*)pb;
            b0.q[1] = *(const f32x4*)(pb + 16);
            pb += 16 * BS_STRIDE;
            b1.q[0] = *(const f32x4*)pb;
            b1.q[1] = *(const f32x4*)(pb + 16);
        }
        c[0][0] = __builtin_amdgcn_wmma_f32_16x16x32_f16(false, a0.v, false, b0.v, (short)0, c[0][0], false, false);
        c[0][1] = __builtin_amdgcn_wmma_f32_16x16x32_f16(false, a0.v, false, b1.v, (short)0, c[0][1], false, false);
        c[1][0] = __builtin_amdgcn_wmma_f32_16x16x32_f16(false, a1.v, false, b0.v, (short)0, c[1][0], false, false);
        c[1][1] = __builtin_amdgcn_wmma_f32_16x16x32_f16(false, a1.v, false, b1.v, (short)0, c[1][1], false, false);

        _Float16* tp;
        tp = Asc; Asc = Asn; Asn = tp;
        tp = Bsc; Bsc = Bsn; Bsn = tp;
    }
}

#define DECL_TILES \
    __shared__ alignas(16) _Float16 As[2][128 * AS_STRIDE]; \
    __shared__ alignas(16) _Float16 Bs[2][64 * BS_STRIDE];

// ---------------------------------------------------------------------------
// Generic GEMM: out = A @ B^T(src N-major) + bias, optional relu.
// Outputs: f32 row-major (outF,ldc), f16 row-major (outH,ldc),
//          f16 transposed (outT,ldcT): outT[col][row] packed 8 rows per store.
// ---------------------------------------------------------------------------
__global__ __launch_bounds__(256) void gemm_kernel(
    const _Float16* __restrict__ A, int lda,
    const _Float16* __restrict__ Bm, int ldb,
    float* __restrict__ outF, _Float16* __restrict__ outH,
    _Float16* __restrict__ outT, int ldc, int ldcT,
    const float* __restrict__ bias, int relu, int K)
{
    DECL_TILES
    const int m0 = blockIdx.y * 128;
    const int n0 = blockIdx.x * 64;
    v8f c[2][2];
    gemm_core(A, lda, Bm, ldb, m0, n0, K, As[0], Bs[0], As[1], Bs[1], c);

    const int t = threadIdx.x;
    const int lane = t & 31;
    const int waveId = t >> 5;
    const int wm = (waveId >> 1) * 32;
    const int wn = (waveId & 1) * 32;
    const int cn = lane & 15;
    const int rOff = (lane < 16) ? 0 : 8;
#pragma unroll
    for (int mt = 0; mt < 2; ++mt)
#pragma unroll
        for (int nt = 0; nt < 2; ++nt) {
            const int gcol = n0 + wn + nt * 16 + cn;
            const int rbase = m0 + wm + mt * 16 + rOff;
            const float bv = bias ? bias[gcol] : 0.f;
            if (outT) {
                union { f32x4 q; _Float16 e[8]; } pk;
#pragma unroll
                for (int r = 0; r < 8; ++r) pk.e[r] = (_Float16)(c[mt][nt][r] + bv);
                *(f32x4*)(outT + (size_t)gcol * ldcT + rbase) = pk.q;
            }
#pragma unroll
            for (int r = 0; r < 8; ++r) {
                float val = c[mt][nt][r] + bv;
                if (relu) val = fmaxf(val, 0.f);
                if (outF) outF[(size_t)(rbase + r) * ldc + gcol] = val;
                if (outH) outH[(size_t)(rbase + r) * ldc + gcol] = (_Float16)val;
            }
        }
}

// ---------------------------------------------------------------------------
// Attention logits: per (b,h) Q·K^T * scale + pad_mask*NEG + NEG*(1-graph_mask)
// K-matrix rows are d-contiguous => already the N-major B source.
// ---------------------------------------------------------------------------
__global__ __launch_bounds__(256) void attn_logits_kernel(
    const _Float16* __restrict__ q16, const _Float16* __restrict__ k16,
    const int* __restrict__ x, const float* __restrict__ gmask,
    float* __restrict__ logits)
{
    DECL_TILES
    const int z = blockIdx.z, b = z >> 3, hh = z & 7;
    const _Float16* A  = q16 + (size_t)b * SSQ * DM + hh * DHD;
    const _Float16* Bm = k16 + (size_t)b * SSQ * DM + hh * DHD;
    const int m0 = blockIdx.y * 128;
    const int n0 = blockIdx.x * 64;
    v8f c[2][2];
    gemm_core(A, DM, Bm, DM, m0, n0, DHD, As[0], Bs[0], As[1], Bs[1], c);

    float* out = logits + (size_t)z * SSQ * SSQ;
    const float* gmb = gmask + (size_t)b * SSQ * SSQ;
    const int* xb = x + b * SSQ;
    const float scale = 0.125f; // 1/sqrt(64)

    const int t = threadIdx.x;
    const int lane = t & 31;
    const int waveId = t >> 5;
    const int wm = (waveId >> 1) * 32;
    const int wn = (waveId & 1) * 32;
    const int cn = lane & 15;
    const int rOff = (lane < 16) ? 0 : 8;
#pragma unroll
    for (int mt = 0; mt < 2; ++mt)
#pragma unroll
        for (int nt = 0; nt < 2; ++nt) {
            const int gcol = n0 + wn + nt * 16 + cn;
            const float pm = (xb[gcol] == PADTOK) ? NEGINF : 0.f;
#pragma unroll
            for (int r = 0; r < 8; ++r) {
                const int grow = m0 + wm + mt * 16 + rOff + r;
                out[(size_t)grow * SSQ + gcol] =
                    c[mt][nt][r] * scale + pm
                    + NEGINF * (1.f - gmb[(size_t)grow * SSQ + gcol]);
            }
        }
}

// ---------------------------------------------------------------------------
// Attention context: per (b,h) ctx = probs @ V, using transposed V
// (vT[D][NB*SSQ]) so B is K-contiguous.  M=1024, N=64, K=1024.
// ---------------------------------------------------------------------------
__global__ __launch_bounds__(256) void attn_ctx_kernel(
    const _Float16* __restrict__ attn16, const _Float16* __restrict__ vT,
    _Float16* __restrict__ ctx16)
{
    DECL_TILES
    const int z = blockIdx.z, b = z >> 3, hh = z & 7;
    const int R = NB * SSQ;
    const _Float16* A  = attn16 + (size_t)z * SSQ * SSQ;
    const _Float16* Bm = vT + (size_t)hh * DHD * R + (size_t)b * SSQ;
    const int m0 = blockIdx.y * 128;
    v8f c[2][2];
    gemm_core(A, SSQ, Bm, R, m0, 0, SSQ, As[0], Bs[0], As[1], Bs[1], c);

    _Float16* out = ctx16 + (size_t)b * SSQ * DM + hh * DHD;
    const int t = threadIdx.x;
    const int lane = t & 31;
    const int waveId = t >> 5;
    const int wm = (waveId >> 1) * 32;
    const int wn = (waveId & 1) * 32;
    const int cn = lane & 15;
    const int rOff = (lane < 16) ? 0 : 8;
#pragma unroll
    for (int mt = 0; mt < 2; ++mt)
#pragma unroll
        for (int nt = 0; nt < 2; ++nt) {
            const int gcol = wn + nt * 16 + cn;
#pragma unroll
            for (int r = 0; r < 8; ++r) {
                const int grow = m0 + wm + mt * 16 + rOff + r;
                out[(size_t)grow * DM + gcol] = (_Float16)c[mt][nt][r];
            }
        }
}

// ---------------------------------------------------------------------------
// Softmax over rows of length 1024 (f32 in, f16 probs out). One block per row.
// ---------------------------------------------------------------------------
__global__ __launch_bounds__(256) void softmax_kernel(
    const float* __restrict__ logits, _Float16* __restrict__ attn)
{
    __shared__ float red[256];
    const size_t base = (size_t)blockIdx.x * SSQ;
    const int t = threadIdx.x;
    float v[4];
    float mx = -3.4e38f;
#pragma unroll
    for (int j = 0; j < 4; ++j) { v[j] = logits[base + t + 256 * j]; mx = fmaxf(mx, v[j]); }
    red[t] = mx; __syncthreads();
    for (int s = 128; s > 0; s >>= 1) { if (t < s) red[t] = fmaxf(red[t], red[t + s]); __syncthreads(); }
    mx = red[0]; __syncthreads();
    float sum = 0.f;
#pragma unroll
    for (int j = 0; j < 4; ++j) { v[j] = __expf(v[j] - mx); sum += v[j]; }
    red[t] = sum; __syncthreads();
    for (int s = 128; s > 0; s >>= 1) { if (t < s) red[t] += red[t + s]; __syncthreads(); }
    const float inv = 1.f / red[0];
#pragma unroll
    for (int j = 0; j < 4; ++j) attn[base + t + 256 * j] = (_Float16)(v[j] * inv);
}

// ---------------------------------------------------------------------------
// Residual + LayerNorm over rows of length 512. One block per row.
// ---------------------------------------------------------------------------
__global__ __launch_bounds__(256) void ln_kernel(
    const float* __restrict__ resid, const float* __restrict__ delta,
    const float* __restrict__ g, const float* __restrict__ bta,
    float* __restrict__ outF, _Float16* __restrict__ outH)
{
    __shared__ float red[256];
    const size_t base = (size_t)blockIdx.x * DM;
    const int t = threadIdx.x;
    float x0 = resid[base + t]       + delta[base + t];
    float x1 = resid[base + t + 256] + delta[base + t + 256];
    red[t] = x0 + x1; __syncthreads();
    for (int s = 128; s > 0; s >>= 1) { if (t < s) red[t] += red[t + s]; __syncthreads(); }
    const float mu = red[0] * (1.f / 512.f);
    __syncthreads();
    const float d0 = x0 - mu, d1 = x1 - mu;
    red[t] = d0 * d0 + d1 * d1; __syncthreads();
    for (int s = 128; s > 0; s >>= 1) { if (t < s) red[t] += red[t + s]; __syncthreads(); }
    const float inv = rsqrtf(red[0] * (1.f / 512.f) + 1e-6f);
    const float y0 = d0 * inv * g[t]       + bta[t];
    const float y1 = d1 * inv * g[t + 256] + bta[t + 256];
    outF[base + t]       = y0;
    outF[base + t + 256] = y1;
    outH[base + t]       = (_Float16)y0;
    outH[base + t + 256] = (_Float16)y1;
}

// ---------------------------------------------------------------------------
// Embedding * sqrt(D) + positional encoding -> h (f32) and h16 (f16)
// ---------------------------------------------------------------------------
__global__ __launch_bounds__(256) void embed_kernel(
    const int* __restrict__ x, const float* __restrict__ emb,
    float* __restrict__ h, _Float16* __restrict__ h16)
{
    const int idx = blockIdx.x * 256 + threadIdx.x;   // < 8192*512 exactly
    const int row = idx >> 9;
    const int i   = idx & 511;
    const int s   = row & (SSQ - 1);
    const int tok = x[row];
    const float rate = expf(-(float)(2 * (i >> 1)) * (9.210340371976184f / 512.0f));
    const float ar = (float)s * rate;
    const float pe = ((i & 1) == 0) ? sinf(ar) : cosf(ar);
    const float val = emb[tok * DM + i] * 22.62741699796952f + pe; // sqrt(512)
    h[idx]   = val;
    h16[idx] = (_Float16)val;
}

// ---------------------------------------------------------------------------
// f32 -> f16 weight conversion with per-layer transpose:
// dst[l][n][k] = src[l][k][n].  dst-linear indexing => coalesced writes.
// ---------------------------------------------------------------------------
__global__ void cvt_t_kernel(const float* __restrict__ src, _Float16* __restrict__ dst,
                             int L, int K, int N)
{
    const int i = blockIdx.x * 256 + threadIdx.x;
    const int slice = K * N;
    if (i < L * slice) {
        const int l = i / slice;
        const int r = i - l * slice;
        const int n = r / K;
        const int k = r - n * K;
        dst[i] = (_Float16)src[(size_t)l * slice + (size_t)k * N + n];
    }
}

// ---------------------------------------------------------------------------
// Output head: sigmoid(h @ Wout + bout). One wave per row, shfl reduction.
// ---------------------------------------------------------------------------
__global__ __launch_bounds__(256) void out_kernel(
    const float* __restrict__ h, const float* __restrict__ Wout,
    const float* __restrict__ bout, float* __restrict__ out)
{
    const int t = threadIdx.x;
    const int lane = t & 31;
    const int row = blockIdx.x * 8 + (t >> 5);
    const float* hp = h + (size_t)row * DM;
    float s = 0.f;
#pragma unroll
    for (int j = 0; j < 16; ++j) s += hp[lane + 32 * j] * Wout[lane + 32 * j];
#pragma unroll
    for (int off = 16; off > 0; off >>= 1) s += __shfl_xor(s, off, 32);
    if (lane == 0) out[row] = 1.f / (1.f + __expf(-(s + bout[0])));
}

// ---------------------------------------------------------------------------
extern "C" void kernel_launch(void* const* d_in, const int* in_sizes, int n_in,
                              void* d_out, int out_size, void* d_ws, size_t ws_size,
                              hipStream_t stream)
{
    (void)in_sizes; (void)n_in; (void)out_size; (void)ws_size;
    const int*   x    = (const int*)  d_in[0];
    const float* gm   = (const float*)d_in[1];
    const float* emb  = (const float*)d_in[2];
    const float* Wq   = (const float*)d_in[3];
    const float* bq   = (const float*)d_in[4];
    const float* Wk   = (const float*)d_in[5];
    const float* bk   = (const float*)d_in[6];
    const float* Wv   = (const float*)d_in[7];
    const float* bv   = (const float*)d_in[8];
    const float* Wo   = (const float*)d_in[9];
    const float* bo   = (const float*)d_in[10];
    const float* W1   = (const float*)d_in[11];
    const float* b1   = (const float*)d_in[12];
    const float* W2   = (const float*)d_in[13];
    const float* b2   = (const float*)d_in[14];
    const float* ln1g = (const float*)d_in[15];
    const float* ln1b = (const float*)d_in[16];
    const float* ln2g = (const float*)d_in[17];
    const float* ln2b = (const float*)d_in[18];
    const float* Wout = (const float*)d_in[19];
    const float* bout = (const float*)d_in[20];
    float* out = (float*)d_out;

    const size_t R  = (size_t)NB * SSQ;   // 8192 rows
    const size_t ZZ = (size_t)NB * NH;    // 64 (b,h) pairs

    char* p = (char*)d_ws;
    auto take = [&](size_t bytes) -> char* {
        char* r = p; p += (bytes + 255) & ~(size_t)255; return r;
    };
    float*    h      = (float*)   take(R * DM * 4);
    _Float16* h16    = (_Float16*)take(R * DM * 2);
    _Float16* q16    = (_Float16*)take(R * DM * 2);
    _Float16* k16    = (_Float16*)take(R * DM * 2);
    _Float16* vT     = (_Float16*)take((size_t)DM * R * 2);   // [D][B*S]
    _Float16* ctx16  = (_Float16*)take(R * DM * 2);
    float*    tmp    = (float*)   take(R * DM * 4);
    _Float16* ffn16  = (_Float16*)take(R * FF * 2);
    float*    logits = (float*)   take(ZZ * SSQ * SSQ * 4);
    _Float16* attn16 = (_Float16*)take(ZZ * SSQ * SSQ * 2);
    _Float16* wqT    = (_Float16*)take((size_t)LAYERS * DM * DM * 2);
    _Float16* wkT    = (_Float16*)take((size_t)LAYERS * DM * DM * 2);
    _Float16* wvT    = (_Float16*)take((size_t)LAYERS * DM * DM * 2);
    _Float16* woT    = (_Float16*)take((size_t)LAYERS * DM * DM * 2);
    _Float16* w1T    = (_Float16*)take((size_t)LAYERS * DM * FF * 2);
    _Float16* w2T    = (_Float16*)take((size_t)LAYERS * FF * DM * 2);

    // weight conversion + transpose (every call; deterministic)
    const int nW = LAYERS * DM * DM;
    const int nF = LAYERS * DM * FF;
    cvt_t_kernel<<<(nW + 255) / 256, 256, 0, stream>>>(Wq, wqT, LAYERS, DM, DM);
    cvt_t_kernel<<<(nW + 255) / 256, 256, 0, stream>>>(Wk, wkT, LAYERS, DM, DM);
    cvt_t_kernel<<<(nW + 255) / 256, 256, 0, stream>>>(Wv, wvT, LAYERS, DM, DM);
    cvt_t_kernel<<<(nW + 255) / 256, 256, 0, stream>>>(Wo, woT, LAYERS, DM, DM);
    cvt_t_kernel<<<(nF + 255) / 256, 256, 0, stream>>>(W1, w1T, LAYERS, DM, FF);
    cvt_t_kernel<<<(nF + 255) / 256, 256, 0, stream>>>(W2, w2T, LAYERS, FF, DM);

    embed_kernel<<<(int)(R * DM / 256), 256, 0, stream>>>(x, emb, h, h16);

    const dim3 gProj(DM / 64, (int)(R / 128));     // (8, 64)
    const dim3 gF1(FF / 64, (int)(R / 128));       // (32, 64)
    const dim3 gLog(SSQ / 64, SSQ / 128, (int)ZZ); // (16, 8, 64)
    const dim3 gCtx(1, SSQ / 128, (int)ZZ);        // (1, 8, 64)

    for (int l = 0; l < LAYERS; ++l) {
        const _Float16* wq_l = wqT + (size_t)l * DM * DM;
        const _Float16* wk_l = wkT + (size_t)l * DM * DM;
        const _Float16* wv_l = wvT + (size_t)l * DM * DM;
        const _Float16* wo_l = woT + (size_t)l * DM * DM;
        const _Float16* w1_l = w1T + (size_t)l * DM * FF;
        const _Float16* w2_l = w2T + (size_t)l * FF * DM;

        // Q,K (f16 row-major), V (f16 transposed [D][R])
        gemm_kernel<<<gProj, 256, 0, stream>>>(h16, DM, wq_l, DM, nullptr, q16, nullptr, DM, 0, bq + l * DM, 0, DM);
        gemm_kernel<<<gProj, 256, 0, stream>>>(h16, DM, wk_l, DM, nullptr, k16, nullptr, DM, 0, bk + l * DM, 0, DM);
        gemm_kernel<<<gProj, 256, 0, stream>>>(h16, DM, wv_l, DM, nullptr, nullptr, vT, DM, (int)R, bv + l * DM, 0, DM);

        // attention
        attn_logits_kernel<<<gLog, 256, 0, stream>>>(q16, k16, x, gm, logits);
        softmax_kernel<<<(int)(ZZ * SSQ), 256, 0, stream>>>(logits, attn16);
        attn_ctx_kernel<<<gCtx, 256, 0, stream>>>(attn16, vT, ctx16);

        // output projection (f32) + residual LN1
        gemm_kernel<<<gProj, 256, 0, stream>>>(ctx16, DM, wo_l, DM, tmp, nullptr, nullptr, DM, 0, bo + l * DM, 0, DM);
        ln_kernel<<<(int)R, 256, 0, stream>>>(h, tmp, ln1g + l * DM, ln1b + l * DM, h, h16);

        // FFN: relu(h1@W1+b1) (f16 only) then @W2+b2 (f32), residual LN2
        gemm_kernel<<<gF1, 256, 0, stream>>>(h16, DM, w1_l, DM, nullptr, ffn16, nullptr, FF, 0, b1 + l * FF, 1, DM);
        gemm_kernel<<<gProj, 256, 0, stream>>>(ffn16, FF, w2_l, FF, tmp, nullptr, nullptr, DM, 0, b2 + l * DM, 0, FF);
        ln_kernel<<<(int)R, 256, 0, stream>>>(h, tmp, ln2g + l * DM, ln2b + l * DM, h, h16);
    }

    out_kernel<<<(int)(R / 8), 256, 0, stream>>>(h, Wout, bout, out);
}
